// MambaWrap_85289460564657
// MI455X (gfx1250) — compile-verified, matrix operand order checked by reading, not run on hardware
//
#include <hip/hip_runtime.h>
#include <hip/hip_bf16.h>
#include <math.h>

// ---------------------------------------------------------------------------
// Mamba-2 block + 5x5 conv tail for MI455X (gfx1250, wave32, WMMA).
// GEMMs (in_proj / out_proj / conv2d-as-implicit-GEMM) use
// v_wmma_f32_16x16x32_bf16 with fp32 accumulation; 32x64 tile per wave
// (2 A fragments x 4 B subtiles = 8 accumulators) to halve B-panel traffic.
// Sequential selective scan runs one workgroup per (batch, head).
// Workspace requirement: ~354 MB.
// ---------------------------------------------------------------------------

typedef __attribute__((ext_vector_type(16))) __bf16 v16bf;
typedef __attribute__((ext_vector_type(8)))  float  v8f;

#define DM      768          // D_MODEL
#define DI      1536         // D_INNER
#define DSTATE  128
#define NH      24           // NHEADS
#define HD      64           // HEADDIM
#define CDIM    1792         // CONV_DIM
#define DPROJ   3352         // D_IN_PROJ
#define NPAD    3392         // D_IN_PROJ padded to multiple of 64
#define BATCH   2
#define LSEQ    4352
#define IMG     4096
#define HEADL   256          // LSEQ - IMG
#define ROWS    (BATCH*LSEQ) // 8704
#define EPSF    1e-5f
#define PIMG    68           // 64 + 2*2 padded image extent

static __device__ __forceinline__ float sigmoidf_(float x){ return 1.f/(1.f+expf(-x)); }

#define WMMA_BF16(ACC,AV,BV) \
  ACC = __builtin_amdgcn_wmma_f32_16x16x32_bf16(false,(AV),false,(BV),(short)0,(ACC),false,false)

// ------------------------------ weight prep --------------------------------
__global__ void k_prep_inw(const float* __restrict__ w, __bf16* __restrict__ o){
  int idx = blockIdx.x*256 + threadIdx.x;
  if (idx >= NPAD*DM) return;
  int n = idx / DM, k = idx % DM;
  o[idx] = (__bf16)(n < DPROJ ? w[(size_t)k*DPROJ + n] : 0.f);
}
__global__ void k_prep_outw(const float* __restrict__ w, __bf16* __restrict__ o){
  int idx = blockIdx.x*256 + threadIdx.x;
  if (idx >= DM*DI) return;
  int n = idx / DI, k = idx % DI;          // o[n][k] <- w[k][n]
  o[idx] = (__bf16)w[(size_t)k*DM + n];
}
__global__ void k_prep_convw(const float* __restrict__ w, __bf16* __restrict__ o){
  size_t idx = (size_t)blockIdx.x*256 + threadIdx.x;
  if (idx >= (size_t)25*DM*DM) return;
  int ic = idx % DM; size_t r = idx / DM;
  int oc = r % DM;  int tap = r / DM;
  int ky = tap/5, kx = tap%5;              // w[oc][ic][ky][kx]
  o[idx] = (__bf16)w[(((size_t)oc*DM + ic)*5 + ky)*5 + kx];
}
__global__ void k_zero_img(__bf16* __restrict__ img){
  size_t idx = (size_t)blockIdx.x*256 + threadIdx.x;
  if (idx < (size_t)BATCH*PIMG*PIMG*DM) img[idx] = (__bf16)0.f;
}

// ------------------------------ layernorms ---------------------------------
__global__ __launch_bounds__(256)
void k_ln1(const float* __restrict__ x, const float* __restrict__ w,
           const float* __restrict__ bb, __bf16* __restrict__ out){
  int row = blockIdx.x, tid = threadIdx.x;
  __shared__ float red[256];
  const float* xr = x + (size_t)row*DM;
  float v0 = xr[tid], v1 = xr[tid+256], v2 = xr[tid+512];
  red[tid] = v0+v1+v2; __syncthreads();
  for (int s=128; s>0; s>>=1){ if(tid<s) red[tid]+=red[tid+s]; __syncthreads(); }
  float mean = red[0]*(1.f/768.f); __syncthreads();
  float d0=v0-mean,d1=v1-mean,d2=v2-mean;
  red[tid]=d0*d0+d1*d1+d2*d2; __syncthreads();
  for (int s=128; s>0; s>>=1){ if(tid<s) red[tid]+=red[tid+s]; __syncthreads(); }
  float rs = rsqrtf(red[0]*(1.f/768.f) + EPSF);
  __bf16* o = out + (size_t)row*DM;
  o[tid]     = (__bf16)(d0*rs*w[tid]     + bb[tid]);
  o[tid+256] = (__bf16)(d1*rs*w[tid+256] + bb[tid+256]);
  o[tid+512] = (__bf16)(d2*rs*w[tid+512] + bb[tid+512]);
}

__global__ __launch_bounds__(256)
void k_ln2img(const float* __restrict__ hbuf, const float* __restrict__ w,
              const float* __restrict__ bb, __bf16* __restrict__ img){
  int row = blockIdx.x, tid = threadIdx.x;   // row in [0, BATCH*IMG)
  int b = row >> 12, pix = row & 4095;
  __shared__ float red[256];
  const float* xr = hbuf + ((size_t)(b*LSEQ + HEADL + pix))*DM;
  float v0 = xr[tid], v1 = xr[tid+256], v2 = xr[tid+512];
  red[tid] = v0+v1+v2; __syncthreads();
  for (int s=128; s>0; s>>=1){ if(tid<s) red[tid]+=red[tid+s]; __syncthreads(); }
  float mean = red[0]*(1.f/768.f); __syncthreads();
  float d0=v0-mean,d1=v1-mean,d2=v2-mean;
  red[tid]=d0*d0+d1*d1+d2*d2; __syncthreads();
  for (int s=128; s>0; s>>=1){ if(tid<s) red[tid]+=red[tid+s]; __syncthreads(); }
  float rs = rsqrtf(red[0]*(1.f/768.f) + EPSF);
  __bf16* o = img + ((size_t)((b*PIMG + 2 + (pix>>6))*PIMG + 2 + (pix&63)))*DM;
  o[tid]     = (__bf16)(d0*rs*w[tid]     + bb[tid]);
  o[tid+256] = (__bf16)(d1*rs*w[tid+256] + bb[tid+256]);
  o[tid+512] = (__bf16)(d2*rs*w[tid+512] + bb[tid+512]);
}

// ---- generic 32x64 wave-tile GEMM body: C[m0..m0+31][n0..n0+63] ----------
// A row-major [M][K] bf16 ; W packed [N][K] bf16 ; C row-major [M][ldc] f32.
template<int K, int LDC>
static __device__ __forceinline__
void gemm32x64(const __bf16* __restrict__ A, const __bf16* __restrict__ W,
               float* __restrict__ C, int m0, int n0, int lane){
  int kh = (lane>>4)*16;
  const __bf16* ap0 = A + (size_t)(m0      + (lane&15))*K + kh;
  const __bf16* ap1 = A + (size_t)(m0 + 16 + (lane&15))*K + kh;
  const __bf16* bp  = W + (size_t)(n0      + (lane&15))*K + kh;
  v8f c00={},c01={},c02={},c03={},c10={},c11={},c12={},c13={};
  for (int k=0;k<K;k+=32){
    v16bf a0 = *(const v16bf*)(ap0 + k);
    v16bf a1 = *(const v16bf*)(ap1 + k);
    v16bf b0 = *(const v16bf*)(bp + k);
    v16bf b1 = *(const v16bf*)(bp + (size_t)16*K + k);
    v16bf b2 = *(const v16bf*)(bp + (size_t)32*K + k);
    v16bf b3 = *(const v16bf*)(bp + (size_t)48*K + k);
    WMMA_BF16(c00,a0,b0); WMMA_BF16(c01,a0,b1);
    WMMA_BF16(c02,a0,b2); WMMA_BF16(c03,a0,b3);
    WMMA_BF16(c10,a1,b0); WMMA_BF16(c11,a1,b1);
    WMMA_BF16(c12,a1,b2); WMMA_BF16(c13,a1,b3);
  }
  int lh = lane>>4, nn = lane&15;
  #pragma unroll
  for (int r=0;r<8;++r){
    size_t m = (size_t)m0 + r + 8*lh;
    C[m*LDC + n0      + nn] = c00[r];
    C[m*LDC + n0 + 16 + nn] = c01[r];
    C[m*LDC + n0 + 32 + nn] = c02[r];
    C[m*LDC + n0 + 48 + nn] = c03[r];
    m += 16;
    C[m*LDC + n0      + nn] = c10[r];
    C[m*LDC + n0 + 16 + nn] = c11[r];
    C[m*LDC + n0 + 32 + nn] = c12[r];
    C[m*LDC + n0 + 48 + nn] = c13[r];
  }
}

// ------------------- GEMM: in_proj  [ROWS,768] x [768,NPAD] ----------------
__global__ __launch_bounds__(32)
void k_gemm_inproj(const __bf16* __restrict__ A, const __bf16* __restrict__ W,
                   float* __restrict__ C){
  gemm32x64<DM,NPAD>(A, W, C, blockIdx.x*32, blockIdx.y*64, threadIdx.x);
}

// ------------------- GEMM: out_proj [ROWS,1536] x [1536,768] ---------------
__global__ __launch_bounds__(32)
void k_gemm_outproj(const __bf16* __restrict__ A, const __bf16* __restrict__ W,
                    float* __restrict__ C){
  gemm32x64<DI,DM>(A, W, C, blockIdx.x*32, blockIdx.y*64, threadIdx.x);
}

// --------- conv2d 5x5 SAME as implicit GEMM, fused bias + residual ---------
__global__ __launch_bounds__(32)
void k_conv2d(const __bf16* __restrict__ img, const __bf16* __restrict__ W,
              const float* __restrict__ xin, const float* __restrict__ bias,
              float* __restrict__ out){
  int b  = blockIdx.z;
  int m0 = blockIdx.x*32;                 // 32-pixel tile inside one 64-wide row
  int n0 = blockIdx.y*64;
  int lane = threadIdx.x;
  int kh = (lane>>4)*16;
  int py = m0 >> 6, px0 = m0 & 63;
  v8f c00={},c01={},c02={},c03={},c10={},c11={},c12={},c13={};
  for (int tap=0; tap<25; ++tap){
    int ky = tap/5, kx = tap%5;
    const __bf16* ap0 = img + ((size_t)((b*PIMG + py + ky)*PIMG + px0      + kx + (lane&15)))*DM + kh;
    const __bf16* ap1 = img + ((size_t)((b*PIMG + py + ky)*PIMG + px0 + 16 + kx + (lane&15)))*DM + kh;
    const __bf16* bp  = W + ((size_t)tap*DM + n0 + (lane&15))*DM + kh;
    for (int k=0;k<DM;k+=32){
      v16bf a0 = *(const v16bf*)(ap0 + k);
      v16bf a1 = *(const v16bf*)(ap1 + k);
      v16bf b0 = *(const v16bf*)(bp + k);
      v16bf b1 = *(const v16bf*)(bp + (size_t)16*DM + k);
      v16bf b2 = *(const v16bf*)(bp + (size_t)32*DM + k);
      v16bf b3 = *(const v16bf*)(bp + (size_t)48*DM + k);
      WMMA_BF16(c00,a0,b0); WMMA_BF16(c01,a0,b1);
      WMMA_BF16(c02,a0,b2); WMMA_BF16(c03,a0,b3);
      WMMA_BF16(c10,a1,b0); WMMA_BF16(c11,a1,b1);
      WMMA_BF16(c12,a1,b2); WMMA_BF16(c13,a1,b3);
    }
  }
  int lh = lane>>4, nn = lane&15;
  #pragma unroll
  for (int r=0;r<8;++r){
    #pragma unroll
    for (int half=0; half<2; ++half){
      int pix = m0 + half*16 + r + 8*lh;
      size_t o = ((size_t)(b*LSEQ + HEADL + pix))*DM;
      const v8f* cc0 = half ? &c10 : &c00;
      const v8f* cc1 = half ? &c11 : &c01;
      const v8f* cc2 = half ? &c12 : &c02;
      const v8f* cc3 = half ? &c13 : &c03;
      int n;
      n = n0      + nn; out[o+n] = (*cc0)[r] + bias[n] + xin[o+n];
      n = n0 + 16 + nn; out[o+n] = (*cc1)[r] + bias[n] + xin[o+n];
      n = n0 + 32 + nn; out[o+n] = (*cc2)[r] + bias[n] + xin[o+n];
      n = n0 + 48 + nn; out[o+n] = (*cc3)[r] + bias[n] + xin[o+n];
    }
  }
}

// --------------------------- conv1d + SiLU ---------------------------------
__global__ void k_conv1d(const float* __restrict__ zx, const float* __restrict__ w,
                         const float* __restrict__ bias, float* __restrict__ xBCc){
  size_t idx = (size_t)blockIdx.x*256 + threadIdx.x;
  if (idx >= (size_t)ROWS*CDIM) return;
  int c = idx % CDIM; size_t r = idx / CDIM;
  int l = r % LSEQ, b = r / LSEQ;
  float s = bias[c];
  #pragma unroll
  for (int k=0;k<4;++k){
    int ls = l + k - 3;
    if (ls >= 0) s += w[c*4+k] * zx[((size_t)(b*LSEQ+ls))*NPAD + DI + c];
  }
  xBCc[idx] = s * sigmoidf_(s);
}

// ----------------------------- dt / dA -------------------------------------
__global__ void k_dtda(const float* __restrict__ zx, const float* __restrict__ dtb,
                       const float* __restrict__ A_log, float* __restrict__ dt,
                       float* __restrict__ dA){
  int idx = blockIdx.x*256 + threadIdx.x;
  if (idx >= ROWS*NH) return;
  int h = idx % NH; size_t r = idx / NH;
  float x = zx[r*NPAD + (DI + CDIM) + h] + dtb[h];
  float d = x > 0.f ? x + log1pf(expf(-x)) : log1pf(expf(x));   // softplus
  dt[idx] = d;
  dA[idx] = expf(d * (-expf(A_log[h])));
}

// --------------------------- selective scan --------------------------------
__global__ __launch_bounds__(256)
void k_scan(const float* __restrict__ xBCc, const float* __restrict__ dt,
            const float* __restrict__ dA, const float* __restrict__ Dskip,
            float* __restrict__ ybuf){
  int bh = blockIdx.x;
  int b = bh / NH, h = bh % NH;
  int tid = threadIdx.x;
  int p = tid & 63, q = tid >> 6;          // p: headdim row, q: state quarter
  __shared__ float xsh[64], Bsh[128], Csh[128], part[256];
  float hst[32];
  #pragma unroll
  for (int i=0;i<32;++i) hst[i]=0.f;
  float Dh = Dskip[h];
  for (int l=0;l<LSEQ;++l){
    size_t row = (size_t)b*LSEQ + l;
    const float* xr = xBCc + row*CDIM;
    if (tid < 64)  xsh[tid] = xr[h*HD + tid];
    if (tid < 128) Bsh[tid] = xr[DI + tid];
    else           Csh[tid-128] = xr[DI + DSTATE + (tid-128)];
    float dtv = dt[row*NH + h];
    float dAv = dA[row*NH + h];
    __syncthreads();
    float coef = dtv * xsh[p];
    const float* Bq = Bsh + q*32;
    const float* Cq = Csh + q*32;
    float acc = 0.f;
    #pragma unroll
    for (int i=0;i<32;++i){ hst[i] = hst[i]*dAv + coef*Bq[i]; acc += hst[i]*Cq[i]; }
    part[tid] = acc;
    __syncthreads();
    if (tid < 64){
      float y = part[tid] + part[64+tid] + part[128+tid] + part[192+tid] + Dh*xsh[tid];
      ybuf[row*DI + h*HD + tid] = y;
    }
  }
}

// -------------------- gating (SiLU(z)) + RMS norm --------------------------
__global__ __launch_bounds__(256)
void k_gate(const float* __restrict__ zx, const float* __restrict__ ybuf,
            const float* __restrict__ nw, __bf16* __restrict__ ybar){
  int row = blockIdx.x, tid = threadIdx.x;
  __shared__ float red[256];
  const float* zr = zx + (size_t)row*NPAD;
  const float* yr = ybuf + (size_t)row*DI;
  float g[6]; float ss = 0.f;
  #pragma unroll
  for (int i=0;i<6;++i){
    int c = tid + i*256;
    float z = zr[c];
    float gi = yr[c] * (z * sigmoidf_(z));
    g[i] = gi; ss += gi*gi;
  }
  red[tid] = ss; __syncthreads();
  for (int s=128; s>0; s>>=1){ if(tid<s) red[tid]+=red[tid+s]; __syncthreads(); }
  float rs = rsqrtf(red[0]*(1.f/1536.f) + EPSF);
  __bf16* o = ybar + (size_t)row*DI;
  #pragma unroll
  for (int i=0;i<6;++i){ int c = tid + i*256; o[c] = (__bf16)(g[i]*rs*nw[c]); }
}

// -------------------- residual for non-image head rows ---------------------
__global__ void k_head(const float* __restrict__ x, const float* __restrict__ hbuf,
                       float* __restrict__ out){
  int idx = blockIdx.x*256 + threadIdx.x;
  if (idx >= BATCH*HEADL*DM) return;
  int d = idx % DM; int r = idx / DM;      // r in [0, 512)
  size_t o = ((size_t)((r/HEADL)*LSEQ + (r%HEADL)))*DM + d;
  out[o] = x[o] + hbuf[o];
}

// ---------------------------------------------------------------------------
extern "C" void kernel_launch(void* const* d_in, const int* in_sizes, int n_in,
                              void* d_out, int out_size, void* d_ws, size_t ws_size,
                              hipStream_t stream){
  const float* x      = (const float*)d_in[0];
  const float* ln1_w  = (const float*)d_in[1];
  const float* ln1_b  = (const float*)d_in[2];
  const float* inw    = (const float*)d_in[3];
  const float* c1w    = (const float*)d_in[4];
  const float* c1b    = (const float*)d_in[5];
  const float* dtb    = (const float*)d_in[6];
  const float* A_log  = (const float*)d_in[7];
  const float* Dskip  = (const float*)d_in[8];
  const float* norm_w = (const float*)d_in[9];
  const float* outw   = (const float*)d_in[10];
  const float* ln2_w  = (const float*)d_in[11];
  const float* ln2_b  = (const float*)d_in[12];
  const float* c2w    = (const float*)d_in[13];
  const float* c2b    = (const float*)d_in[14];
  float* out = (float*)d_out;

  // workspace layout (bytes, all 256-aligned; total ~354 MB)
  char* ws = (char*)d_ws;
  size_t off = 0;
  __bf16* wb_in   = (__bf16*)(ws + off); off += (size_t)NPAD*DM*2;        // 5.2 MB
  __bf16* wb_out  = (__bf16*)(ws + off); off += (size_t)DM*DI*2;          // 2.4 MB
  __bf16* wb_conv = (__bf16*)(ws + off); off += (size_t)25*DM*DM*2;       // 29.5 MB
  __bf16* ubf     = (__bf16*)(ws + off); off += (size_t)ROWS*DM*2;        // 13.4 MB
  float*  zx      = (float*) (ws + off); off += (size_t)ROWS*NPAD*4;      // 118 MB
  float*  xBCc    = (float*) (ws + off); off += (size_t)ROWS*CDIM*4;      // 62.4 MB
  float*  dt      = (float*) (ws + off); off += (size_t)ROWS*NH*4;        // 0.8 MB
  float*  dA      = (float*) (ws + off); off += (size_t)ROWS*NH*4;        // 0.8 MB
  float*  ybuf    = (float*) (ws + off); off += (size_t)ROWS*DI*4;        // 53.5 MB
  __bf16* ybar    = (__bf16*)(ws + off); off += (size_t)ROWS*DI*2;        // 26.7 MB
  float*  hbuf    = (float*) (ws + off); off += (size_t)ROWS*DM*4;        // 26.7 MB
  __bf16* imgb    = (__bf16*)(ws + off); off += (size_t)BATCH*PIMG*PIMG*DM*2; // 14.2 MB

  // 1) weight repack to [N][K] bf16
  k_prep_inw  <<<(NPAD*DM+255)/256, 256, 0, stream>>>(inw, wb_in);
  k_prep_outw <<<(DM*DI+255)/256,   256, 0, stream>>>(outw, wb_out);
  k_prep_convw<<<(int)(((size_t)25*DM*DM+255)/256), 256, 0, stream>>>(c2w, wb_conv);

  // 2) LN1 -> bf16 activations
  k_ln1<<<ROWS, 256, 0, stream>>>(x, ln1_w, ln1_b, ubf);

  // 3) in_proj GEMM (WMMA, 32x64 wave tiles)
  k_gemm_inproj<<<dim3(ROWS/32, NPAD/64), 32, 0, stream>>>(ubf, wb_in, zx);

  // 4) depthwise causal conv1d + SiLU ; dt/dA
  k_conv1d<<<(int)(((size_t)ROWS*CDIM+255)/256), 256, 0, stream>>>(zx, c1w, c1b, xBCc);
  k_dtda<<<(ROWS*NH+255)/256, 256, 0, stream>>>(zx, dtb, A_log, dt, dA);

  // 5) selective scan (one WG per (b,h))
  k_scan<<<BATCH*NH, 256, 0, stream>>>(xBCc, dt, dA, Dskip, ybuf);

  // 6) gate + RMS norm -> bf16
  k_gate<<<ROWS, 256, 0, stream>>>(zx, ybuf, norm_w, ybar);

  // 7) out_proj GEMM (WMMA, 32x64 wave tiles)
  k_gemm_outproj<<<dim3(ROWS/32, DM/64), 32, 0, stream>>>(ybar, wb_out, hbuf);

  // 8) head rows: residual only
  k_head<<<(BATCH*HEADL*DM+255)/256, 256, 0, stream>>>(x, hbuf, out);

  // 9) LN2 on image tail into zero-padded bf16 image
  k_zero_img<<<(int)(((size_t)BATCH*PIMG*PIMG*DM+255)/256), 256, 0, stream>>>(imgb);
  k_ln2img<<<BATCH*IMG, 256, 0, stream>>>(hbuf, ln2_w, ln2_b, imgb);

  // 10) conv2d implicit GEMM (WMMA), fused bias + residual, writes d_out tail
  k_conv2d<<<dim3(IMG/32, DM/64, BATCH), 32, 0, stream>>>(imgb, wb_conv, x, c2b, out);
}
